// MViT_10273561772252
// MI455X (gfx1250) — matrix operand order repaired and checked by reading, not compiled, not run on hardware
//
#include <hip/hip_runtime.h>

typedef __attribute__((ext_vector_type(16))) __bf16 v16bf;
typedef __attribute__((ext_vector_type(8)))  float  v8f;

struct __align__(16) U4 { unsigned int x, y, z, w; };
union BFV { v16bf v; unsigned int u[8]; U4 q[2]; };

#define NBATCH 8
#define NHEADS 8
#define NTOK   3137
#define CDIM   768
#define DDIM   96
#define NPOOL  785
#define M_QKV  (NBATCH*NTOK)      // 25096
#define M_PROJ (NBATCH*NPOOL)     // 6280
#define SCALE_QK 0.1020620726159658f

static __device__ inline unsigned short f2bf(float f) {
    union { float f; unsigned int u; } x; x.f = f;
    unsigned int u = x.u;
    unsigned int r = (u + 0x7FFFu + ((u >> 16) & 1u)) >> 16;
    return (unsigned short)r;
}
static __device__ inline float bf2f(unsigned short h) {
    union { unsigned int u; float f; } x; x.u = ((unsigned int)h) << 16;
    return x.f;
}
static __device__ inline v8f vzero8() {
    v8f z = {0.f,0.f,0.f,0.f,0.f,0.f,0.f,0.f};
    return z;
}

// ---------------- f32 -> bf16 conversion ----------------
__global__ void cvt_bf16_kernel(const float* __restrict__ s,
                                unsigned short* __restrict__ d, int n) {
    for (int i = blockIdx.x * blockDim.x + threadIdx.x; i < n;
         i += gridDim.x * blockDim.x)
        d[i] = f2bf(s[i]);
}

// ---------------- generic bf16 WMMA GEMM: out = A @ W^T + bias ----------------
// A: M x 768 row-major bf16 (as uints), W: Nout x 768 row-major bf16 (as uints)
// Each wave computes a 32M x 64N tile (2 A-tiles x 4 B-tiles = 8 WMMAs / k-step)
// so every B load feeds two WMMAs (~21 FLOP per cache byte).
// mode 0: fp32 row-major out. mode 1: scatter bf16 into q/k/v (b,h,n,d) buffers.
__global__ __launch_bounds__(128) void gemm_bf16_kernel(
    const unsigned int* __restrict__ A, const unsigned int* __restrict__ W,
    const float* __restrict__ bias, int M, int Nout, int mode,
    float* __restrict__ outF, unsigned short* __restrict__ qb,
    unsigned short* __restrict__ kb, unsigned short* __restrict__ vb) {
    const int tid  = threadIdx.x;
    const int wv   = tid >> 5;
    const int lane = tid & 31;
    const int half = lane >> 4;
    const int l15  = lane & 15;
    const int mbase = blockIdx.x * 128 + wv * 32;
    const int nbase = blockIdx.y * 64;

    int arow0 = mbase + l15;        if (arow0 >= M) arow0 = M - 1;
    int arow1 = mbase + 16 + l15;   if (arow1 >= M) arow1 = M - 1;
    const size_t aoff0 = (size_t)arow0 * 48;   // 48 uints = 96 bf16 per row
    const size_t aoff1 = (size_t)arow1 * 48;

    v8f acc[2][4];
#pragma unroll
    for (int mt = 0; mt < 2; mt++)
#pragma unroll
        for (int c = 0; c < 4; c++) acc[mt][c] = vzero8();

#pragma unroll 2
    for (int ks = 0; ks < 24; ks++) {
        BFV au0, au1;
        au0.q[0] = *reinterpret_cast<const U4*>(A + aoff0 + 16 * ks + 4 * half);
        au0.q[1] = *reinterpret_cast<const U4*>(A + aoff0 + 16 * ks + 4 * half + 8);
        au1.q[0] = *reinterpret_cast<const U4*>(A + aoff1 + 16 * ks + 4 * half);
        au1.q[1] = *reinterpret_cast<const U4*>(A + aoff1 + 16 * ks + 4 * half + 8);
        if (ks + 1 < 24) {  // hide next A fetch latency (global_prefetch_b8)
            __builtin_prefetch(A + aoff0 + 16 * (ks + 1) + 4 * half, 0, 1);
            __builtin_prefetch(A + aoff1 + 16 * (ks + 1) + 4 * half, 0, 1);
        }
#pragma unroll
        for (int c = 0; c < 4; c++) {
            const int ncol = nbase + c * 16 + l15;
            const size_t boff = (size_t)ncol * 48 + 16 * ks + 4 * half;
            BFV bu;
            bu.q[0] = *reinterpret_cast<const U4*>(W + boff);
            bu.q[1] = *reinterpret_cast<const U4*>(W + boff + 8);
            acc[0][c] = __builtin_amdgcn_wmma_f32_16x16x32_bf16(
                false, au0.v, false, bu.v, (short)0, acc[0][c], false, false);
            acc[1][c] = __builtin_amdgcn_wmma_f32_16x16x32_bf16(
                false, au1.v, false, bu.v, (short)0, acc[1][c], false, false);
        }
    }

#pragma unroll
    for (int mt = 0; mt < 2; mt++) {
#pragma unroll
        for (int c = 0; c < 4; c++) {
            const int ncol = nbase + c * 16 + l15;
            const float bb = bias[ncol];
#pragma unroll
            for (int i = 0; i < 8; i++) {
                const int row = mbase + mt * 16 + i + 8 * half;
                if (row >= M) continue;
                const float val = acc[mt][c][i] + bb;
                if (mode == 0) {
                    outF[(size_t)row * Nout + ncol] = val;
                } else {
                    const int s   = ncol / CDIM;
                    const int rem = ncol - s * CDIM;
                    const int hd  = rem / DDIM;
                    const int dm  = rem - hd * DDIM;
                    const int b_  = row / NTOK;
                    const int n_  = row - b_ * NTOK;
                    const size_t di =
                        (((size_t)b_ * NHEADS + hd) * NTOK + n_) * DDIM + dm;
                    unsigned short h16 = f2bf(val);
                    if (s == 0) qb[di] = h16;
                    else if (s == 1) kb[di] = h16;
                    else vb[di] = h16;
                }
            }
        }
    }
}

// ---------------- depthwise 3x3x3 pool (stride 1,2,2) + LayerNorm ----------------
__global__ __launch_bounds__(128) void pool_ln_kernel(
    const unsigned short* __restrict__ src, const float* __restrict__ wconv,
    const float* __restrict__ g, const float* __restrict__ bb,
    unsigned short* __restrict__ dst) {
    const int pix = blockIdx.x % NPOOL;
    const int bh  = blockIdx.x / NPOOL;
    const int tid = threadIdx.x;
    const bool act = tid < DDIM;
    const int dm = tid;

    float val = 0.f;
    if (act) {
        if (pix == 0) {
            val = bf2f(src[((size_t)bh * NTOK) * DDIM + dm]);
        } else {
            const int sp = pix - 1;
            const int wp = sp % 14;
            const int hp = (sp / 14) % 14;
            const int tp = sp / 196;
            float a = 0.f;
#pragma unroll
            for (int dt = 0; dt < 3; dt++)
#pragma unroll
                for (int dh = 0; dh < 3; dh++)
#pragma unroll
                    for (int dw = 0; dw < 3; dw++) {
                        const int t  = tp + dt - 1;
                        const int hh = 2 * hp + dh - 1;
                        const int ww = 2 * wp + dw - 1;
                        if (t >= 0 && t < 4 && hh >= 0 && hh < 28 &&
                            ww >= 0 && ww < 28) {
                            const int n = 1 + (t * 28 + hh) * 28 + ww;
                            a += bf2f(src[((size_t)bh * NTOK + n) * DDIM + dm]) *
                                 wconv[dm * 27 + (dt * 3 + dh) * 3 + dw];
                        }
                    }
            val = a;
        }
    }

    __shared__ float red[128];
    red[tid] = act ? val : 0.f;
    __syncthreads();
#pragma unroll
    for (int s = 64; s > 0; s >>= 1) {
        if (tid < s) red[tid] += red[tid + s];
        __syncthreads();
    }
    const float mean = red[0] / (float)DDIM;
    __syncthreads();
    const float dvi = act ? (val - mean) : 0.f;
    red[tid] = dvi * dvi;
    __syncthreads();
#pragma unroll
    for (int s = 64; s > 0; s >>= 1) {
        if (tid < s) red[tid] += red[tid + s];
        __syncthreads();
    }
    const float var = red[0] / (float)DDIM;

    if (act) {
        const float nrm = (val - mean) * rsqrtf(var + 1e-5f) * g[dm] + bb[dm];
        dst[((size_t)bh * NPOOL + pix) * DDIM + dm] = f2bf(nrm);
    }
}

// ---------------- flash attention over pooled tokens (bf16 WMMA) ----------------
#define KPADU 50   // K_lds row stride in uints (100 bf16)
#define VPAD  40   // Vlds (transposed) row stride in bf16

__global__ __launch_bounds__(128) void attn_kernel(
    const unsigned short* __restrict__ qp, const unsigned short* __restrict__ kp,
    const unsigned short* __restrict__ vp, unsigned short* __restrict__ obf) {
    const int bh   = blockIdx.x;
    const int b    = bh >> 3;
    const int head = bh & 7;
    const int tid  = threadIdx.x;
    const int wv   = tid >> 5;
    const int lane = tid & 31;
    const int half = lane >> 4;
    const int l15  = lane & 15;
    const int qbase = blockIdx.y * 64 + wv * 16;

    __shared__ unsigned int   Klds[32 * KPADU];
    __shared__ unsigned short Vlds[DDIM * VPAD];
    __shared__ unsigned short Plds[4][16][34];

    // Preload the three Q A-operands (16 rows x 96 dims) for this wave.
    const unsigned int* qpu = reinterpret_cast<const unsigned int*>(qp);
    int qrow = qbase + l15;
    if (qrow > NPOOL - 1) qrow = NPOOL - 1;
    const size_t qoff = ((size_t)bh * NPOOL + qrow) * 48;
    BFV aq[3];
#pragma unroll
    for (int ks = 0; ks < 3; ks++) {
        aq[ks].q[0] = *reinterpret_cast<const U4*>(qpu + qoff + 16 * ks + 4 * half);
        aq[ks].q[1] = *reinterpret_cast<const U4*>(qpu + qoff + 16 * ks + 4 * half + 8);
    }

    v8f accO[6];
#pragma unroll
    for (int c = 0; c < 6; c++) accO[c] = vzero8();
    float mrun[8], lrun[8];
#pragma unroll
    for (int i = 0; i < 8; i++) { mrun[i] = -3.0e38f; lrun[i] = 0.f; }

    const unsigned int* kpu = reinterpret_cast<const unsigned int*>(kp);

    for (int kbase = 0; kbase < 800; kbase += 32) {
        __syncthreads();
        // Stage K: 32 keys x 96 bf16, row stride 100 bf16 (conflict-free reads).
        for (int idx = tid; idx < 1536; idx += 128) {
            const int kl = idx / 48, u = idx - kl * 48;
            const int kg = kbase + kl;
            Klds[kl * KPADU + u] =
                (kg < NPOOL) ? kpu[((size_t)bh * NPOOL + kg) * 48 + u] : 0u;
        }
        // Stage V transposed: Vlds[dim][key], row stride 40 bf16.
        for (int idx = tid; idx < 3072; idx += 128) {
            const int kl = idx / 96, dm = idx - kl * 96;
            const int kg = kbase + kl;
            Vlds[dm * VPAD + kl] =
                (kg < NPOOL) ? vp[((size_t)bh * NPOOL + kg) * DDIM + dm]
                             : (unsigned short)0;
        }
        // Prefetch next key block's K and V rows while we compute this one.
        {
            const int nb = kbase + 32;
            const int pr = tid >> 2;        // 32 rows
            const int sg = tid & 3;         // 4 segments of 48B
            const int kg = nb + pr;
            if (nb < NPOOL && kg < NPOOL) {
                const size_t rb = ((size_t)bh * NPOOL + kg) * 96 + sg * 24;
                __builtin_prefetch(kp + rb, 0, 1);
                __builtin_prefetch(vp + rb, 0, 1);
            }
        }
        __syncthreads();

        // S = Q @ K^T for two 16-key tiles.
        v8f stile[2];
#pragma unroll
        for (int t = 0; t < 2; t++) {
            v8f s = vzero8();
            const int key = t * 16 + l15;
#pragma unroll
            for (int ks = 0; ks < 3; ks++) {
                BFV bu;
                const int ub = key * KPADU + ks * 16 + 4 * half;
#pragma unroll
                for (int u = 0; u < 8; u++) bu.u[u] = Klds[ub + (u < 4 ? u : u + 4)];
                s = __builtin_amdgcn_wmma_f32_16x16x32_bf16(
                    false, aq[ks].v, false, bu.v, (short)0, s, false, false);
            }
            stile[t] = s;
        }

        // scale + mask invalid keys
#pragma unroll
        for (int t = 0; t < 2; t++) {
            const bool valid = (kbase + t * 16 + l15) < NPOOL;
#pragma unroll
            for (int i = 0; i < 8; i++) {
                const float v = stile[t][i];
                stile[t][i] = valid ? v * SCALE_QK : -1e30f;
            }
        }

        // online softmax (rows live across 16 lanes of one half)
#pragma unroll
        for (int i = 0; i < 8; i++) {
            const float t0 = stile[0][i], t1 = stile[1][i];
            float mx = fmaxf(t0, t1);
#pragma unroll
            for (int off = 1; off < 16; off <<= 1)
                mx = fmaxf(mx, __shfl_xor(mx, off, 32));
            const float mnew  = fmaxf(mrun[i], mx);
            const float alpha = __expf(mrun[i] - mnew);
            const float p0 = __expf(t0 - mnew);
            const float p1 = __expf(t1 - mnew);
            float rs = p0 + p1;
#pragma unroll
            for (int off = 1; off < 16; off <<= 1)
                rs += __shfl_xor(rs, off, 32);
            lrun[i] = lrun[i] * alpha + rs;
            mrun[i] = mnew;
#pragma unroll
            for (int c = 0; c < 6; c++) accO[c][i] *= alpha;
            const int r = i + 8 * half;
            Plds[wv][r][l15]      = f2bf(p0);
            Plds[wv][r][16 + l15] = f2bf(p1);
        }
        asm volatile("s_wait_dscnt 0" ::: "memory");

        // Re-layout P (C-layout in LDS) into a 16x32 A operand.
        BFV ap;
#pragma unroll
        for (int p = 0; p < 8; p++) {
            const int koff = ((p < 4) ? p * 2 : 16 + (p - 4) * 2) + 8 * half;
            ap.u[p] = reinterpret_cast<const unsigned int*>(&Plds[wv][l15][0])[koff >> 1];
        }

        // O += P @ V  (B read from transposed V: contiguous per lane)
#pragma unroll
        for (int c = 0; c < 6; c++) {
            BFV bv_;
            const int dm = c * 16 + l15;
#pragma unroll
            for (int u = 0; u < 8; u++)
                bv_.u[u] = reinterpret_cast<const unsigned int*>(Vlds)[dm * (VPAD / 2) + 4 * half + (u < 4 ? u : u + 4)];
            accO[c] = __builtin_amdgcn_wmma_f32_16x16x32_bf16(
                false, ap.v, false, bv_.v, (short)0, accO[c], false, false);
        }
    }

    // epilogue: normalize and store O in (b, nq, C) layout (bf16 for proj GEMM)
#pragma unroll
    for (int i = 0; i < 8; i++) {
        const int row = qbase + i + 8 * half;
        const float inv = (lrun[i] > 0.f) ? 1.f / lrun[i] : 0.f;
        if (row < NPOOL) {
#pragma unroll
            for (int c = 0; c < 6; c++) {
                const float v = accO[c][i] * inv;
                obf[((size_t)b * NPOOL + row) * CDIM + head * DDIM + c * 16 + l15] =
                    f2bf(v);
            }
        }
    }
}

// ---------------- host side ----------------
extern "C" void kernel_launch(void* const* d_in, const int* in_sizes, int n_in,
                              void* d_out, int out_size, void* d_ws, size_t ws_size,
                              hipStream_t stream) {
    (void)in_sizes; (void)n_in; (void)out_size; (void)ws_size;
    const float* x      = (const float*)d_in[0];
    const float* W_qkv  = (const float*)d_in[1];
    const float* b_qkv  = (const float*)d_in[2];
    const float* W_proj = (const float*)d_in[3];
    const float* b_proj = (const float*)d_in[4];
    const float* wq = (const float*)d_in[5];
    const float* wk = (const float*)d_in[6];
    const float* wvv = (const float*)d_in[7];
    const float* gq = (const float*)d_in[8];
    const float* bq = (const float*)d_in[9];
    const float* gk = (const float*)d_in[10];
    const float* bk = (const float*)d_in[11];
    const float* gv = (const float*)d_in[12];
    const float* bv = (const float*)d_in[13];

    char* ws = (char*)d_ws;
    size_t off = 0;
    auto alloc = [&](size_t bytes) {
        char* p = ws + off;
        off += (bytes + 255) & ~(size_t)255;
        return p;
    };
    const size_t XBF_N   = (size_t)M_QKV * CDIM;          // 19,273,728
    const size_t WQKV_N  = (size_t)3 * CDIM * CDIM;       // 1,769,472
    const size_t WPROJ_N = (size_t)CDIM * CDIM;           // 589,824
    const size_t QKV_N   = (size_t)NBATCH * NHEADS * NTOK * DDIM;
    const size_t POOL_N  = (size_t)NBATCH * NHEADS * NPOOL * DDIM;
    const size_t O_N     = (size_t)NBATCH * NPOOL * CDIM;

    unsigned short* XBF   = (unsigned short*)alloc(XBF_N * 2);
    unsigned short* WQKVB = (unsigned short*)alloc(WQKV_N * 2);
    unsigned short* WPROJB= (unsigned short*)alloc(WPROJ_N * 2);
    unsigned short* QBF   = (unsigned short*)alloc(QKV_N * 2);
    unsigned short* KBF   = (unsigned short*)alloc(QKV_N * 2);
    unsigned short* VBF   = (unsigned short*)alloc(QKV_N * 2);
    unsigned short* QP    = (unsigned short*)alloc(POOL_N * 2);
    unsigned short* KP    = (unsigned short*)alloc(POOL_N * 2);
    unsigned short* VP    = (unsigned short*)alloc(POOL_N * 2);
    unsigned short* OBF   = (unsigned short*)alloc(O_N * 2);

    cvt_bf16_kernel<<<2048, 256, 0, stream>>>(x, XBF, (int)XBF_N);
    cvt_bf16_kernel<<<512, 256, 0, stream>>>(W_qkv, WQKVB, (int)WQKV_N);
    cvt_bf16_kernel<<<256, 256, 0, stream>>>(W_proj, WPROJB, (int)WPROJ_N);

    // QKV GEMM: (25096 x 768) @ (2304 x 768)^T -> scatter bf16 q/k/v
    gemm_bf16_kernel<<<dim3((M_QKV + 127) / 128, 3 * CDIM / 64), 128, 0, stream>>>(
        (const unsigned int*)XBF, (const unsigned int*)WQKVB, b_qkv,
        M_QKV, 3 * CDIM, 1, nullptr, QBF, KBF, VBF);

    // attention pool + LN for q, k, v
    const int poolBlocks = NBATCH * NHEADS * NPOOL;
    pool_ln_kernel<<<poolBlocks, 128, 0, stream>>>(QBF, wq, gq, bq, QP);
    pool_ln_kernel<<<poolBlocks, 128, 0, stream>>>(KBF, wk, gk, bk, KP);
    pool_ln_kernel<<<poolBlocks, 128, 0, stream>>>(VBF, wvv, gv, bv, VP);

    // flash attention -> O (b, nq, C) bf16
    attn_kernel<<<dim3(NBATCH * NHEADS, (NPOOL + 63) / 64), 128, 0, stream>>>(
        QP, KP, VP, OBF);

    // proj GEMM: (6280 x 768) @ (768 x 768)^T + bias -> fp32 d_out
    gemm_bf16_kernel<<<dim3((M_PROJ + 127) / 128, CDIM / 64), 128, 0, stream>>>(
        (const unsigned int*)OBF, (const unsigned int*)WPROJB, b_proj,
        M_PROJ, CDIM, 0, (float*)d_out, nullptr, nullptr, nullptr);
}
